// MambaEncoder_4887672783590
// MI455X (gfx1250) — matrix operand order repaired and checked by reading, not compiled, not run on hardware
//
#include <hip/hip_runtime.h>
#include <hip/hip_bf16.h>
#include <math.h>

#define LAYERS 2
#define D_IN 128
#define D_MODEL 256
#define D_INNER 512
#define D_STATE 16
#define D_CONV 4
#define DT_RANK 16
#define NCLS 6
#define BB 8
#define LL 1024
#define M_ROWS (BB * LL) /* 8192 */

typedef __attribute__((ext_vector_type(16))) __bf16 v16bf;
typedef __attribute__((ext_vector_type(8)))  float  v8f;
typedef __attribute__((ext_vector_type(4)))  unsigned v4u;
typedef __attribute__((ext_vector_type(8)))  int      v8i_;
typedef __attribute__((ext_vector_type(4)))  int      v4i_;

__device__ __forceinline__ float4 ld4(const float* p) {
  return *reinterpret_cast<const float4*>(p);
}
__device__ __forceinline__ void pack8(v16bf& v, int base, float4 lo, float4 hi) {
  v[base + 0] = (__bf16)lo.x; v[base + 1] = (__bf16)lo.y;
  v[base + 2] = (__bf16)lo.z; v[base + 3] = (__bf16)lo.w;
  v[base + 4] = (__bf16)hi.x; v[base + 5] = (__bf16)hi.y;
  v[base + 6] = (__bf16)hi.z; v[base + 7] = (__bf16)hi.w;
}
__device__ __forceinline__ v8f zero8f() {
  v8f v;
#pragma unroll
  for (int i = 0; i < 8; ++i) v[i] = 0.0f;
  return v;
}
__device__ __forceinline__ float sigmoidf_(float x) {
  return 1.0f / (1.0f + __expf(-x));
}

// ---------------------------------------------------------------------------
// WMMA GEMM: C[M,N] = A[M,K] * W[N,K]^T (+ bias[N])
// A, W row-major f32; converted in-register to bf16; f32 accumulate.
// Each wave owns a 16x64 output tile (4 sub-tiles of 16x16), K stepped by 32.
// K is a template constant (128/256/512); M % 16 == 0 guaranteed by callers.
// BOUNDED: N not a multiple of 64 -> B loads use clamped (valid) addresses
// for out-of-range columns (their outputs are never stored), so the inner
// loop stays branch-free with EXEC all-1s at every WMMA.
// ---------------------------------------------------------------------------
template <int K, bool BOUNDED>
__global__ __launch_bounds__(256) void wmma_gemm_kernel(
    const float* __restrict__ A, const float* __restrict__ W,
    const float* __restrict__ bias, float* __restrict__ C,
    int M, int N, int tilesN, int totalWaves, int useBias) {
  int gw = (blockIdx.x * blockDim.x + threadIdx.x) >> 5;
  if (gw >= totalWaves) return;  // wave-uniform: EXEC all-1 inside

  const int lane = threadIdx.x & 31;
  const int hi16 = lane >> 4;   // 0: lanes 0-15, 1: lanes 16-31
  const int r16  = lane & 15;
  const int tm = gw / tilesN;
  const int tn = gw % tilesN;
  const int mBase = tm * 16;
  const int nBase = tn * 64;

  v8f acc0 = zero8f(), acc1 = zero8f(), acc2 = zero8f(), acc3 = zero8f();

  const float* Arow = A + (size_t)(mBase + r16) * K;

  // Column for each B sub-tile; clamped for loads when BOUNDED.
  int col[4];
  const float* Wp[4];
#pragma unroll
  for (int s = 0; s < 4; ++s) {
    col[s] = nBase + s * 16 + r16;
    int cl = BOUNDED ? (col[s] < N ? col[s] : N - 1) : col[s];
    Wp[s] = W + (size_t)cl * K + hi16 * 16;
  }

#pragma unroll 4
  for (int kk = 0; kk < K; kk += 32) {
    if (kk + 32 < K) __builtin_prefetch(Arow + kk + 32, 0, 1);
    // A operand (16x32 bf16): elems 0..15 = K[kk+hi16*16+0..15]
    const int k0 = kk + hi16 * 16;
    v16bf av;
    pack8(av, 0, ld4(Arow + k0), ld4(Arow + k0 + 4));
    pack8(av, 8, ld4(Arow + k0 + 8), ld4(Arow + k0 + 12));

    // B operands (32x16 bf16): lane col, K = kk + hi16*16 + 0..15
    v16bf b0, b1, b2, b3;
    pack8(b0, 0, ld4(Wp[0] + kk), ld4(Wp[0] + kk + 4));
    pack8(b0, 8, ld4(Wp[0] + kk + 8), ld4(Wp[0] + kk + 12));
    pack8(b1, 0, ld4(Wp[1] + kk), ld4(Wp[1] + kk + 4));
    pack8(b1, 8, ld4(Wp[1] + kk + 8), ld4(Wp[1] + kk + 12));
    pack8(b2, 0, ld4(Wp[2] + kk), ld4(Wp[2] + kk + 4));
    pack8(b2, 8, ld4(Wp[2] + kk + 8), ld4(Wp[2] + kk + 12));
    pack8(b3, 0, ld4(Wp[3] + kk), ld4(Wp[3] + kk + 4));
    pack8(b3, 8, ld4(Wp[3] + kk + 8), ld4(Wp[3] + kk + 12));

    acc0 = __builtin_amdgcn_wmma_f32_16x16x32_bf16(false, av, false, b0, (short)0, acc0, false, false);
    acc1 = __builtin_amdgcn_wmma_f32_16x16x32_bf16(false, av, false, b1, (short)0, acc1, false, false);
    acc2 = __builtin_amdgcn_wmma_f32_16x16x32_bf16(false, av, false, b2, (short)0, acc2, false, false);
    acc3 = __builtin_amdgcn_wmma_f32_16x16x32_bf16(false, av, false, b3, (short)0, acc3, false, false);
  }

  // store: C/D layout => lane col = r16, vgpr i row = i + hi16*8
  v8f acc[4] = {acc0, acc1, acc2, acc3};
#pragma unroll
  for (int s = 0; s < 4; ++s) {
    if (BOUNDED && col[s] >= N) continue;
    const float bv = useBias ? bias[col[s]] : 0.0f;
#pragma unroll
    for (int i = 0; i < 8; ++i) {
      const int row = mBase + hi16 * 8 + i;
      C[(size_t)row * N + col[s]] = acc[s][i] + bv;
    }
  }
}

// ---------------------------------------------------------------------------
// Causal depthwise conv (kernel 4) + bias + SiLU on first half of xz.
// ---------------------------------------------------------------------------
__global__ __launch_bounds__(256) void conv_silu_kernel(
    const float* __restrict__ xz, const float* __restrict__ cw,
    const float* __restrict__ cb, float* __restrict__ xi, int total) {
  int idx = blockIdx.x * blockDim.x + threadIdx.x;
  if (idx >= total) return;
  const int d  = idx & (D_INNER - 1);
  const int bl = idx >> 9;           // b*L + l
  const int l  = bl & (LL - 1);
  const float w0 = cw[d * 4 + 0], w1 = cw[d * 4 + 1];
  const float w2 = cw[d * 4 + 2], w3 = cw[d * 4 + 3];
  float acc = cb[d];
  if (l >= 3) acc += xz[(size_t)(bl - 3) * (2 * D_INNER) + d] * w0;
  if (l >= 2) acc += xz[(size_t)(bl - 2) * (2 * D_INNER) + d] * w1;
  if (l >= 1) acc += xz[(size_t)(bl - 1) * (2 * D_INNER) + d] * w2;
  acc += xz[(size_t)bl * (2 * D_INNER) + d] * w3;
  xi[idx] = acc * sigmoidf_(acc);  // silu
}

// ---------------------------------------------------------------------------
// dt = softplus(xdbl[:, :16] @ dpw^T + dpb); xdbl row = 48 floats.
// ---------------------------------------------------------------------------
__global__ __launch_bounds__(512) void dt_softplus_kernel(
    const float* __restrict__ xdbl, const float* __restrict__ dpw,
    const float* __restrict__ dpb, float* __restrict__ dtf) {
  const int m = blockIdx.x;
  const int d = threadIdx.x;
  __shared__ float dtr[DT_RANK];
  if (d < DT_RANK) dtr[d] = xdbl[(size_t)m * 48 + d];
  __syncthreads();
  float acc = dpb[d];
#pragma unroll
  for (int r = 0; r < DT_RANK; ++r) acc += dtr[r] * dpw[d * DT_RANK + r];
  const float sp = (acc > 20.0f) ? acc : log1pf(__expf(acc));
  dtf[(size_t)m * D_INNER + d] = sp;
}

// ---------------------------------------------------------------------------
// Selective scan, fused with D-skip and SiLU(z) gate.
// Lane layout: lane = n + 16*(d&1); each wave handles 2 channels (b, 2d).
// ---------------------------------------------------------------------------
__global__ __launch_bounds__(256) void scan_kernel(
    const float* __restrict__ dtf, const float* __restrict__ xi,
    const float* __restrict__ xdbl, const float* __restrict__ xz,
    const float* __restrict__ Alog, const float* __restrict__ Dpv,
    float* __restrict__ yg) {
  const int gid  = blockIdx.x * blockDim.x + threadIdx.x;
  const int gw   = gid >> 5;
  const int lane = threadIdx.x & 31;
  const int hi16 = lane >> 4;
  const int n    = lane & 15;
  const int b    = gw >> 8;           // 256 wave-slots per batch
  const int pair = gw & 255;
  const int d    = pair * 2 + hi16;

  const float Acoef = -__expf(Alog[d * D_STATE + n]);
  const float Dv = Dpv[d];
  float hc = 0.0f;
  const size_t rowBase = (size_t)b * LL;

  for (int l = 0; l < LL; ++l) {
    const size_t r = rowBase + l;
    const float dt = dtf[r * D_INNER + d];
    const float xv = xi[r * D_INNER + d];
    const float Bn = xdbl[r * 48 + DT_RANK + n];
    const float Cn = xdbl[r * 48 + DT_RANK + D_STATE + n];
    const float dA = __expf(dt * Acoef);
    hc = dA * hc + (dt * xv) * Bn;
    float c = hc * Cn;
    c += __shfl_xor(c, 1, 16);
    c += __shfl_xor(c, 2, 16);
    c += __shfl_xor(c, 4, 16);
    c += __shfl_xor(c, 8, 16);
    if (n == 0) {
      const float z = xz[r * (2 * D_INNER) + D_INNER + d];
      const float sz = z * sigmoidf_(z);
      yg[r * D_INNER + d] = (c + xv * Dv) * sz;
    }
  }
}

// ---------------------------------------------------------------------------
// Mean over L + SE gate. se_w1 (32x256 f32 = 32KB) is staged into LDS by the
// Tensor Data Mover: one wave issues tensor_load_to_lds with a hand-packed D#
// (2-D, data_size=4B, tile == tensor), waits on TENSORcnt, then all waves use
// the LDS copy.
// ---------------------------------------------------------------------------
__global__ __launch_bounds__(256) void mean_se_kernel(
    const float* __restrict__ h, const float* __restrict__ w1,
    const float* __restrict__ b1, const float* __restrict__ w2,
    const float* __restrict__ b2, float* __restrict__ g) {
  const int b = blockIdx.x;
  const int t = threadIdx.x;
  __shared__ float w1s[(D_MODEL / 8) * D_MODEL];  // 32*256 floats = 32KB
  __shared__ float mm[D_MODEL];
  __shared__ float tt[D_MODEL / 8];

#if __has_builtin(__builtin_amdgcn_tensor_load_to_lds)
  if (t < 32) {  // one wave drives the TDM
    const unsigned ldsOff = (unsigned)(size_t)(void*)w1s;  // LDS byte offset = flat addr[31:0]
    const unsigned long long ga = (unsigned long long)(size_t)w1;
    v4u g0;
    g0[0] = 1u;                                   // count=1 (valid descriptor)
    g0[1] = ldsOff;                               // lds_addr
    g0[2] = (unsigned)(ga & 0xffffffffu);         // global_addr[31:0]
    g0[3] = (unsigned)((ga >> 32) & 0x01ffffffu)  // global_addr[56:32]
            | (2u << 30);                         // type=2 ("image")
    v8i_ g1;
    g1[0] = 0x20000;     // data_size=2 (4B), workgroup_mask=0
    g1[1] = 0x01000000;  // tensor_dim0[15:0]=256 in bits 63:48
    g1[2] = 0x00200000;  // tensor_dim1[15:0]=32 in bits 95:80 (bits 111:80)
    g1[3] = 0x01000000;  // tile_dim0=256 in bits 127:112
    g1[4] = 32;          // tile_dim1=32 (bits 143:128)
    g1[5] = 256;         // tensor_dim0_stride[31:0]=256 (bits 191:160)
    g1[6] = 0;
    g1[7] = 0;
    v4i_ z4;
    z4[0] = 0; z4[1] = 0; z4[2] = 0; z4[3] = 0;
#if __clang_major__ >= 23
    v8i_ z8;
    z8[0] = 0; z8[1] = 0; z8[2] = 0; z8[3] = 0;
    z8[4] = 0; z8[5] = 0; z8[6] = 0; z8[7] = 0;
    __builtin_amdgcn_tensor_load_to_lds(g0, g1, z4, z4, z8, 0);
#else
    __builtin_amdgcn_tensor_load_to_lds(g0, g1, z4, z4, 0);
#endif
    __builtin_amdgcn_s_wait_tensorcnt(0);
  }
#else
  for (int i = t; i < (D_MODEL / 8) * D_MODEL; i += 256) w1s[i] = w1[i];
#endif

  const float* hb = h + (size_t)b * LL * D_MODEL;
  float acc = 0.0f;
  for (int l = 0; l < LL; ++l) acc += hb[(size_t)l * D_MODEL + t];
  mm[t] = acc * (1.0f / (float)LL);
  __syncthreads();

  if (t < D_MODEL / 8) {
    float a = b1[t];
    for (int dm = 0; dm < D_MODEL; ++dm) a += mm[dm] * w1s[t * D_MODEL + dm];
    tt[t] = fmaxf(a, 0.0f);
  }
  __syncthreads();
  float a2 = b2[t];
#pragma unroll
  for (int j = 0; j < D_MODEL / 8; ++j) a2 += tt[j] * w2[t * (D_MODEL / 8) + j];
  g[b * D_MODEL + t] = sigmoidf_(a2);
}

// ---------------------------------------------------------------------------
// LiteAttention pooling + head, h gated by g.
// ---------------------------------------------------------------------------
__global__ __launch_bounds__(256) void attn_pool_head_kernel(
    const float* __restrict__ h, const float* __restrict__ g,
    const float* __restrict__ attn_w, const float* __restrict__ attn_b,
    const float* __restrict__ head_w, const float* __restrict__ head_b,
    float* __restrict__ out) {
  const int b = blockIdx.x;
  const int tid = threadIdx.x;
  const int wave = tid >> 5;
  const int lane = tid & 31;
  __shared__ float s[LL];
  __shared__ float red[8];
  __shared__ float gs[D_MODEL];
  __shared__ float pooled[D_MODEL];

  gs[tid] = g[b * D_MODEL + tid];
  __syncthreads();

  const float ab = attn_b[0];
  for (int l = wave; l < LL; l += 8) {
    const float* hp = h + ((size_t)b * LL + l) * D_MODEL;
    float partial = 0.0f;
#pragma unroll
    for (int j = 0; j < D_MODEL / 32; ++j) {
      const int dm = lane + j * 32;
      partial += tanhf(hp[dm] * gs[dm]) * attn_w[dm];
    }
#pragma unroll
    for (int off = 16; off > 0; off >>= 1) partial += __shfl_xor(partial, off, 32);
    if (lane == 0) s[l] = partial + ab;
  }
  __syncthreads();

  // softmax over L
  float lm = -1e30f;
  for (int l = tid; l < LL; l += 256) lm = fmaxf(lm, s[l]);
#pragma unroll
  for (int off = 16; off > 0; off >>= 1) lm = fmaxf(lm, __shfl_xor(lm, off, 32));
  if (lane == 0) red[wave] = lm;
  __syncthreads();
  if (tid == 0) {
    float m = red[0];
#pragma unroll
    for (int w = 1; w < 8; ++w) m = fmaxf(m, red[w]);
    red[0] = m;
  }
  __syncthreads();
  const float mx = red[0];
  __syncthreads();
  float le = 0.0f;
  for (int l = tid; l < LL; l += 256) {
    const float e = __expf(s[l] - mx);
    s[l] = e;
    le += e;
  }
#pragma unroll
  for (int off = 16; off > 0; off >>= 1) le += __shfl_xor(le, off, 32);
  if (lane == 0) red[wave] = le;
  __syncthreads();
  if (tid == 0) {
    float sm = 0.0f;
#pragma unroll
    for (int w = 0; w < 8; ++w) sm += red[w];
    red[0] = 1.0f / sm;
  }
  __syncthreads();
  const float inv = red[0];

  float acc = 0.0f;
  const float* hb = h + (size_t)b * LL * D_MODEL;
  for (int l = 0; l < LL; ++l) acc += s[l] * hb[(size_t)l * D_MODEL + tid];
  pooled[tid] = acc * inv * gs[tid];
  __syncthreads();

  if (tid < NCLS) {
    float o = head_b[tid];
    for (int dm = 0; dm < D_MODEL; ++dm) o += pooled[dm] * head_w[tid * D_MODEL + dm];
    out[b * NCLS + tid] = o;
  }
}

// ---------------------------------------------------------------------------
static inline void launch_gemm(const float* A, const float* W, const float* bias,
                               float* C, int M, int N, int K, int useBias,
                               hipStream_t s) {
  const int tilesN = (N + 63) / 64;
  const int totalWaves = (M / 16) * tilesN;
  const int blocks = (totalWaves * 32 + 255) / 256;
  const bool bounded = (N & 63) != 0;
  if (K == 128) {
    wmma_gemm_kernel<128, false><<<blocks, 256, 0, s>>>(A, W, bias, C, M, N, tilesN, totalWaves, useBias);
  } else if (K == 256) {
    wmma_gemm_kernel<256, false><<<blocks, 256, 0, s>>>(A, W, bias, C, M, N, tilesN, totalWaves, useBias);
  } else if (bounded) {
    wmma_gemm_kernel<512, true><<<blocks, 256, 0, s>>>(A, W, bias, C, M, N, tilesN, totalWaves, useBias);
  } else {
    wmma_gemm_kernel<512, false><<<blocks, 256, 0, s>>>(A, W, bias, C, M, N, tilesN, totalWaves, useBias);
  }
}

extern "C" void kernel_launch(void* const* d_in, const int* in_sizes, int n_in,
                              void* d_out, int out_size, void* d_ws, size_t ws_size,
                              hipStream_t stream) {
  const float* x        = (const float*)d_in[0];
  const float* proj_w   = (const float*)d_in[1];
  const float* proj_b   = (const float*)d_in[2];
  const float* in_proj  = (const float*)d_in[3];
  const float* conv_w   = (const float*)d_in[4];
  const float* conv_b   = (const float*)d_in[5];
  const float* x_proj   = (const float*)d_in[6];
  const float* dt_w     = (const float*)d_in[7];
  const float* dt_b     = (const float*)d_in[8];
  const float* A_log    = (const float*)d_in[9];
  const float* Dp       = (const float*)d_in[10];
  const float* out_proj = (const float*)d_in[11];
  const float* se_w1    = (const float*)d_in[12];
  const float* se_b1    = (const float*)d_in[13];
  const float* se_w2    = (const float*)d_in[14];
  const float* se_b2    = (const float*)d_in[15];
  const float* attn_w   = (const float*)d_in[16];
  const float* attn_b   = (const float*)d_in[17];
  const float* head_w   = (const float*)d_in[18];
  const float* head_b   = (const float*)d_in[19];
  float* out = (float*)d_out;

  // workspace partition (floats); ~98 MB total
  float* W0 = (float*)d_ws;
  float* h0   = W0;
  float* h1   = h0   + (size_t)M_ROWS * D_MODEL;
  float* xz   = h1   + (size_t)M_ROWS * D_MODEL;
  float* xi   = xz   + (size_t)M_ROWS * 2 * D_INNER;
  float* xdbl = xi   + (size_t)M_ROWS * D_INNER;
  float* dtf  = xdbl + (size_t)M_ROWS * 48;
  float* yg   = dtf  + (size_t)M_ROWS * D_INNER;
  float* gbuf = yg   + (size_t)M_ROWS * D_INNER;

  // 1. input projection: h0 = x @ proj_w^T + proj_b
  launch_gemm(x, proj_w, proj_b, h0, M_ROWS, D_MODEL, D_IN, 1, stream);

  const int convTotal = M_ROWS * D_INNER;
  for (int l = 0; l < LAYERS; ++l) {
    const float* h_in  = (l == 0) ? h0 : h1;
    float*       h_out = (l == 0) ? h1 : h0;

    launch_gemm(h_in, in_proj + (size_t)l * 2 * D_INNER * D_MODEL, nullptr, xz,
                M_ROWS, 2 * D_INNER, D_MODEL, 0, stream);

    hipLaunchKernelGGL(conv_silu_kernel, dim3((convTotal + 255) / 256), dim3(256),
                       0, stream, xz, conv_w + (size_t)l * D_INNER * D_CONV,
                       conv_b + (size_t)l * D_INNER, xi, convTotal);

    launch_gemm(xi, x_proj + (size_t)l * 48 * D_INNER, nullptr, xdbl,
                M_ROWS, 48, D_INNER, 0, stream);

    hipLaunchKernelGGL(dt_softplus_kernel, dim3(M_ROWS), dim3(D_INNER), 0, stream,
                       xdbl, dt_w + (size_t)l * D_INNER * DT_RANK,
                       dt_b + (size_t)l * D_INNER, dtf);

    hipLaunchKernelGGL(scan_kernel, dim3(256), dim3(256), 0, stream,
                       dtf, xi, xdbl, xz,
                       A_log + (size_t)l * D_INNER * D_STATE,
                       Dp + (size_t)l * D_INNER, yg);

    launch_gemm(yg, out_proj + (size_t)l * D_MODEL * D_INNER, nullptr, h_out,
                M_ROWS, D_MODEL, D_INNER, 0, stream);
  }

  hipLaunchKernelGGL(mean_se_kernel, dim3(BB), dim3(D_MODEL), 0, stream,
                     h0, se_w1, se_b1, se_w2, se_b2, gbuf);

  hipLaunchKernelGGL(attn_pool_head_kernel, dim3(BB), dim3(256), 0, stream,
                     h0, gbuf, attn_w, attn_b, head_w, head_b, out);
}
